// StackedTransDecoder_81561428951349
// MI455X (gfx1250) — matrix operand order repaired and checked by reading, not compiled, not run on hardware
//
#include <hip/hip_runtime.h>
#include <math.h>

// ---------------- problem constants ----------------
#define B_    128
#define L_    80
#define T_    80
#define E_    512
#define D_    512
#define TWOE_ 1024
#define H_    8
#define DK_   64

typedef __bf16 bf16_t;
typedef __attribute__((ext_vector_type(16))) __bf16 v16bf;
typedef __attribute__((ext_vector_type(8)))  float  v8f;

__device__ __forceinline__ float sigmoidf_(float x) { return 1.0f / (1.0f + __expf(-x)); }

// =====================================================================
// Core bf16 WMMA GEMM:  C(f32)[M,N] = A(bf16)[M,K] @ Wt(bf16)[N,K]^T (+bias) (+C if accum)
// Each wave computes a 16x64 output tile: one A fragment is reused across
// 4 v_wmma_f32_16x16x32_bf16 per K-step (4 f32 accumulators).
// A fragment: row-per-lane, 16 contiguous K per lane half (same k order as B).
// B fragment: col-per-lane, 16 contiguous K per lane half (weights stored N-major).
// =====================================================================
__global__ void k_gemm_bf16(const bf16_t* __restrict__ A,
                            const bf16_t* __restrict__ Wt,
                            const float*  __restrict__ bias,
                            float*        __restrict__ C,
                            bf16_t*       __restrict__ Cb,
                            int M, int N, int K, int accum)
{
    const int lane = threadIdx.x;             // 0..31
    const int wt   = blockIdx.x * blockDim.y + threadIdx.y;  // 64-col tile index
    const int n0   = wt * 64;
    const int m0   = blockIdx.y * 16;
    if (n0 >= N || m0 >= M) return;

    const int half = lane >> 4;               // 0 or 1
    const int lr   = lane & 15;

    const bf16_t* arow = A  + (size_t)(m0 + lr) * K;
    const bf16_t* b0r  = Wt + (size_t)(n0 + lr) * K;
    const bf16_t* b1r  = b0r + (size_t)16 * K;
    const bf16_t* b2r  = b0r + (size_t)32 * K;
    const bf16_t* b3r  = b0r + (size_t)48 * K;

    v8f acc0 = {0.f,0.f,0.f,0.f,0.f,0.f,0.f,0.f};
    v8f acc1 = acc0, acc2 = acc0, acc3 = acc0;

    union Frag { v16bf v; uint4 u[2]; };

    for (int k0 = 0; k0 < K; k0 += 32) {
        Frag a, b0, b1, b2, b3;
        // A 16x32 bf16: lanes 0-15 hold K[0..15]; lanes 16-31 hold K[16..31]
        a.u[0]  = *(const uint4*)(arow + k0 + half * 16);
        a.u[1]  = *(const uint4*)(arow + k0 + half * 16 + 8);
        // B 32x16 bf16 (weights N-major): lane half selects K[0..15] vs K[16..31]
        b0.u[0] = *(const uint4*)(b0r + k0 + half * 16);
        b0.u[1] = *(const uint4*)(b0r + k0 + half * 16 + 8);
        b1.u[0] = *(const uint4*)(b1r + k0 + half * 16);
        b1.u[1] = *(const uint4*)(b1r + k0 + half * 16 + 8);
        b2.u[0] = *(const uint4*)(b2r + k0 + half * 16);
        b2.u[1] = *(const uint4*)(b2r + k0 + half * 16 + 8);
        b3.u[0] = *(const uint4*)(b3r + k0 + half * 16);
        b3.u[1] = *(const uint4*)(b3r + k0 + half * 16 + 8);
        if (k0 + 32 < K) {
            __builtin_prefetch(arow + k0 + 64, 0, 3);
            __builtin_prefetch(b0r  + k0 + 64, 0, 3);
            __builtin_prefetch(b1r  + k0 + 64, 0, 3);
            __builtin_prefetch(b2r  + k0 + 64, 0, 3);
            __builtin_prefetch(b3r  + k0 + 64, 0, 3);
        }
        acc0 = __builtin_amdgcn_wmma_f32_16x16x32_bf16(
                   false, a.v, false, b0.v, (short)0, acc0, false, false);
        acc1 = __builtin_amdgcn_wmma_f32_16x16x32_bf16(
                   false, a.v, false, b1.v, (short)0, acc1, false, false);
        acc2 = __builtin_amdgcn_wmma_f32_16x16x32_bf16(
                   false, a.v, false, b2.v, (short)0, acc2, false, false);
        acc3 = __builtin_amdgcn_wmma_f32_16x16x32_bf16(
                   false, a.v, false, b3.v, (short)0, acc3, false, false);
    }

    const int rbase = m0 + half * 8;
    v8f accs[4] = {acc0, acc1, acc2, acc3};
#pragma unroll
    for (int g = 0; g < 4; ++g) {
        const int col  = n0 + g * 16 + lr;
        const float bv = bias ? bias[col] : 0.f;
        v8f ac = accs[g];
#pragma unroll
        for (int i = 0; i < 8; ++i) {
            size_t off = (size_t)(rbase + i) * N + col;
            float v = ac[i] + bv;
            if (accum) v += C[off];
            C[off] = v;
            if (Cb) Cb[off] = (bf16_t)v;
        }
    }
}

// ---------------- prep kernels ----------------

// src (R x C) f32 row-major -> dst (C x R) bf16  (dst[c*R + r] = src[r*C + c])
__global__ void k_transpose_bf16(const float* __restrict__ src, bf16_t* __restrict__ dst,
                                 int R, int C)
{
    int i = blockIdx.x * 256 + threadIdx.x;
    if (i >= R * C) return;
    int c = i / R, r = i - c * R;
    dst[i] = (bf16_t)src[(size_t)r * C + c];
}

__global__ void k_convert_bf16(const float* __restrict__ s, bf16_t* __restrict__ d, int n)
{
    int i = blockIdx.x * 256 + threadIdx.x;
    if (i < n) d[i] = (bf16_t)s[i];
}

__global__ void k_bias_cat(const float* __restrict__ src, float* __restrict__ dst,
                           int ncopy, int ntotal)
{
    int i = blockIdx.x * 256 + threadIdx.x;
    if (i < ntotal) dst[i] = (i < ncopy) ? src[i] : 0.f;
}

// ysb[t][b][e] = bf16(emb[ys[b,t], e])   (time-major for step access)
__global__ void k_embed(const float* __restrict__ emb, const int* __restrict__ ys,
                        bf16_t* __restrict__ ysb)
{
    int i = blockIdx.x * 256 + threadIdx.x;
    if (i >= T_ * B_ * E_) return;
    int e = i % E_;
    int r = i / E_;
    int b = r % B_;
    int t = r / B_;
    int tok = ys[b * T_ + t];
    ysb[i] = (bf16_t)emb[(size_t)tok * E_ + e];
}

// mean over L with mask; outputs f32 + bf16
__global__ void k_mean(const float* __restrict__ xs_h, const float* __restrict__ xs_mask,
                       float* __restrict__ mf, bf16_t* __restrict__ mb)
{
    int b = blockIdx.x;
    float den = 0.f;
    for (int l = 0; l < L_; ++l) den += xs_mask[b * L_ + l];
    float inv = 1.f / den;
    for (int e = threadIdx.x; e < TWOE_; e += blockDim.x) {
        float s = 0.f;
        for (int l = 0; l < L_; ++l)
            s += xs_h[((size_t)b * L_ + l) * TWOE_ + e] * xs_mask[b * L_ + l];
        float v = s * inv;
        mf[b * TWOE_ + e] = v;
        mb[b * TWOE_ + e] = (bf16_t)v;
    }
}

__global__ void k_tanh_state(const float* __restrict__ in, float* __restrict__ sf,
                             bf16_t* __restrict__ sb, int n)
{
    int i = blockIdx.x * 256 + threadIdx.x;
    if (i >= n) return;
    float v = tanhf(in[i]);
    sf[i] = v;
    sb[i] = (bf16_t)v;
}

// ---------------- per-step elementwise kernels ----------------

// L-GRU combine. g1[B,2560] = [x@Wx + b (2048) | x@Wlin (512)], g2[B,2048] = [h@Wh (1536) | h@Whh (512)]
__global__ void k_lgru_combine(const float* __restrict__ g1, const float* __restrict__ g2,
                               float* __restrict__ sf, bf16_t* __restrict__ sb,
                               const float* __restrict__ ys_mask, int t)
{
    int i = blockIdx.x * 256 + threadIdx.x;
    if (i >= B_ * D_) return;
    int b = i >> 9, j = i & (D_ - 1);
    const float* G1 = g1 + (size_t)b * 2560;
    const float* G2 = g2 + (size_t)b * 2048;
    float r  = sigmoidf_(G1[j]          + G2[j]);
    float z  = sigmoidf_(G1[512 + j]    + G2[512 + j]);
    float lg = sigmoidf_(G1[1024 + j]   + G2[1024 + j]);
    float hc = tanhf(G1[1536 + j] + r * G2[1536 + j] + lg * G1[2048 + j]);
    float h  = sf[i];
    float hn = (1.f - z) * h + z * hc;
    float m  = ys_mask[b * T_ + t];
    float o  = m * hn + (1.f - m) * h;
    sf[i] = o;
    sb[i] = (bf16_t)o;
}

// T-GRU combine. g[B,1536] = [h@Wh + b[:1024] (1024) | h@Whh (512)], bcand = b + 1024
__global__ void k_tgru_combine(const float* __restrict__ g, const float* __restrict__ bcand,
                               float* __restrict__ sf, bf16_t* __restrict__ sb,
                               const float* __restrict__ ys_mask, int t)
{
    int i = blockIdx.x * 256 + threadIdx.x;
    if (i >= B_ * D_) return;
    int b = i >> 9, j = i & (D_ - 1);
    const float* G = g + (size_t)b * 1536;
    float r  = sigmoidf_(G[j]);
    float z  = sigmoidf_(G[512 + j]);
    float hc = tanhf(r * G[1024 + j] + bcand[j]);
    float h  = sf[i];
    float hn = (1.f - z) * h + z * hc;
    float m  = ys_mask[b * T_ + t];
    float o  = m * hn + (1.f - m) * h;
    sf[i] = o;
    sb[i] = (bf16_t)o;
}

// additive multi-head attention: one block per (b,h); 256 threads = 8 waves
__global__ void k_attn(const float* __restrict__ q, const float* __restrict__ uh,
                       const float* __restrict__ xs_h, const float* __restrict__ vvec,
                       const float* __restrict__ xs_mask,
                       float* __restrict__ ctxpre, bf16_t* __restrict__ ctxpre_b)
{
    __shared__ float e_s[L_];
    __shared__ float sinv;
    int b = blockIdx.x / H_;
    int h = blockIdx.x % H_;
    int lane = threadIdx.x & 31;
    int w    = threadIdx.x >> 5;   // 0..7

    // phase 1: scores e[l] = sum_d v[h,d] * tanh(q[b,h,d] + uh[b,l,h,d])
    for (int l = w; l < L_; l += 8) {
        float p = 0.f;
        const float* Q = q  + (size_t)b * D_ + h * DK_;
        const float* U = uh + ((size_t)b * L_ + l) * D_ + h * DK_;
        for (int d = lane; d < DK_; d += 32)
            p += vvec[h * DK_ + d] * tanhf(Q[d] + U[d]);
        for (int off = 16; off; off >>= 1) p += __shfl_xor(p, off, 32);
        if (lane == 0)
            e_s[l] = (xs_mask[b * L_ + l] > 0.f) ? p : -1e9f;
    }
    __syncthreads();

    // phase 2: softmax over l (wave 0)
    if (threadIdx.x < 32) {
        float mx = -1e30f;
        for (int l = lane; l < L_; l += 32) mx = fmaxf(mx, e_s[l]);
        for (int off = 16; off; off >>= 1) mx = fmaxf(mx, __shfl_xor(mx, off, 32));
        float sum = 0.f;
        for (int l = lane; l < L_; l += 32) {
            float a = __expf(e_s[l] - mx);
            e_s[l] = a;
            sum += a;
        }
        for (int off = 16; off; off >>= 1) sum += __shfl_xor(sum, off, 32);
        if (lane == 0) sinv = 1.f / sum;
    }
    __syncthreads();

    // phase 3: ctx_pre[b, h*128 + k] = sum_l alpha[l] * xs_h[b,l,h*128+k]
    const int KC = TWOE_ / H_;  // 128
    if (threadIdx.x < KC) {
        int k = threadIdx.x;
        float acc = 0.f;
        const float* X = xs_h + (size_t)b * L_ * TWOE_ + h * KC + k;
        float si = sinv;
        for (int l = 0; l < L_; ++l)
            acc += e_s[l] * si * X[(size_t)l * TWOE_];
        size_t o = (size_t)b * TWOE_ + h * KC + k;
        ctxpre[o]   = acc;
        ctxpre_b[o] = (bf16_t)acc;
    }
}

// output: out[b,t,j] = tanh(osum[b,j]) * ys_mask[b,t]
__global__ void k_out(const float* __restrict__ osum, const float* __restrict__ ys_mask,
                      float* __restrict__ out, int t)
{
    int i = blockIdx.x * 256 + threadIdx.x;
    if (i >= B_ * D_) return;
    int b = i >> 9, j = i & (D_ - 1);
    out[((size_t)b * T_ + t) * D_ + j] = tanhf(osum[i]) * ys_mask[b * T_ + t];
}

// =====================================================================
// host orchestration
// =====================================================================
extern "C" void kernel_launch(void* const* d_in, const int* in_sizes, int n_in,
                              void* d_out, int out_size, void* d_ws, size_t ws_size,
                              hipStream_t stream)
{
    (void)in_sizes; (void)n_in; (void)out_size; (void)ws_size;

    // ---- inputs (JAX pytree flatten: dict keys sorted alphabetically) ----
    const float* xs_h    = (const float*)d_in[0];
    const int*   ys      = (const int*)  d_in[1];
    const float* xs_mask = (const float*)d_in[2];
    const float* ys_mask = (const float*)d_in[3];
    const float* Wc      = (const float*)d_in[4];
    const float* Wk      = (const float*)d_in[5];
    const float* Ws      = (const float*)d_in[6];
    const float* Ws_init = (const float*)d_in[7];
    const float* Wy      = (const float*)d_in[8];
    const float* a_Wo    = (const float*)d_in[9];
    const float* a_Wq    = (const float*)d_in[10];
    const float* a_v     = (const float*)d_in[11];
    const float* bc      = (const float*)d_in[12];
    const float* bs      = (const float*)d_in[13];
    const float* bs_init = (const float*)d_in[14];
    const float* by      = (const float*)d_in[15];
    const float* cl_Wh   = (const float*)d_in[16];
    const float* cl_Whh  = (const float*)d_in[17];
    const float* cl_Wlin = (const float*)d_in[18];
    const float* cl_Wx   = (const float*)d_in[19];
    const float* cl_b    = (const float*)d_in[20];
    const float* ct_Wh[2]  = { (const float*)d_in[21], (const float*)d_in[24] };
    const float* ct_Whh[2] = { (const float*)d_in[22], (const float*)d_in[25] };
    const float* ct_b[2]   = { (const float*)d_in[23], (const float*)d_in[26] };
    const float* emb     = (const float*)d_in[27];
    const float* lg_Wh   = (const float*)d_in[28];
    const float* lg_Whh  = (const float*)d_in[29];
    const float* lg_Wlin = (const float*)d_in[30];
    const float* lg_Wx   = (const float*)d_in[31];
    const float* lg_b    = (const float*)d_in[32];
    const float* tg_Wh[2]  = { (const float*)d_in[33], (const float*)d_in[36] };
    const float* tg_Whh[2] = { (const float*)d_in[34], (const float*)d_in[37] };
    const float* tg_b[2]   = { (const float*)d_in[35], (const float*)d_in[38] };

    float* out = (float*)d_out;

    // ---- workspace bump allocator ----
    char* wsp = (char*)d_ws;
    size_t off = 0;
    auto alloc = [&](size_t bytes) -> void* {
        void* p = wsp + off;
        off = (off + bytes + 255) & ~(size_t)255;
        return p;
    };

    bf16_t* wk_t     = (bf16_t*)alloc((size_t)D_ * TWOE_ * 2);
    bf16_t* wsinit_t = (bf16_t*)alloc((size_t)D_ * TWOE_ * 2);
    bf16_t* lgru_xw  = (bf16_t*)alloc((size_t)2560 * E_ * 2);      // [Wx^T | Wlin^T], K=512
    bf16_t* lgru_hw  = (bf16_t*)alloc((size_t)2048 * D_ * 2);      // [Wh^T | Whh^T]
    bf16_t* tgru_hw[2];
    tgru_hw[0] = (bf16_t*)alloc((size_t)1536 * D_ * 2);
    tgru_hw[1] = (bf16_t*)alloc((size_t)1536 * D_ * 2);
    bf16_t* cond_xw  = (bf16_t*)alloc((size_t)2560 * TWOE_ * 2);   // K=1024
    bf16_t* cond_hw  = (bf16_t*)alloc((size_t)2048 * D_ * 2);
    bf16_t* ctgru_hw[2];
    ctgru_hw[0] = (bf16_t*)alloc((size_t)1536 * D_ * 2);
    ctgru_hw[1] = (bf16_t*)alloc((size_t)1536 * D_ * 2);
    bf16_t* wq_t = (bf16_t*)alloc((size_t)D_ * D_ * 2);
    bf16_t* wo_t = (bf16_t*)alloc((size_t)TWOE_ * TWOE_ * 2);
    bf16_t* ws_t = (bf16_t*)alloc((size_t)D_ * D_ * 2);
    bf16_t* wy_t = (bf16_t*)alloc((size_t)D_ * E_ * 2);
    bf16_t* wc_t = (bf16_t*)alloc((size_t)D_ * TWOE_ * 2);
    float*  lbias = (float*)alloc(2560 * 4);
    float*  cbias = (float*)alloc(2560 * 4);
    float*  tbias[2]; float* ctbias[2];
    tbias[0]  = (float*)alloc(1536 * 4); tbias[1]  = (float*)alloc(1536 * 4);
    ctbias[0] = (float*)alloc(1536 * 4); ctbias[1] = (float*)alloc(1536 * 4);

    bf16_t* xsb = (bf16_t*)alloc((size_t)B_ * L_ * TWOE_ * 2);
    bf16_t* ysb = (bf16_t*)alloc((size_t)T_ * B_ * E_ * 2);
    float*  uh  = (float*)alloc((size_t)B_ * L_ * D_ * 4);
    float*  meanf = (float*)alloc((size_t)B_ * TWOE_ * 4);
    bf16_t* meanb = (bf16_t*)alloc((size_t)B_ * TWOE_ * 2);
    float*  s0pre = (float*)alloc((size_t)B_ * D_ * 4);
    float*  s_f  = (float*)alloc((size_t)B_ * D_ * 4);
    bf16_t* s_bf = (bf16_t*)alloc((size_t)B_ * D_ * 2);
    float*  g1 = (float*)alloc((size_t)B_ * 2560 * 4);
    float*  g2 = (float*)alloc((size_t)B_ * 2048 * 4);
    float*  tg = (float*)alloc((size_t)B_ * 1536 * 4);
    float*  qb = (float*)alloc((size_t)B_ * D_ * 4);
    float*  ctxpre   = (float*)alloc((size_t)B_ * TWOE_ * 4);
    bf16_t* ctxpre_b = (bf16_t*)alloc((size_t)B_ * TWOE_ * 2);
    float*  ctxf  = (float*)alloc((size_t)B_ * TWOE_ * 4);
    bf16_t* ctx_b = (bf16_t*)alloc((size_t)B_ * TWOE_ * 2);
    float*  osum  = (float*)alloc((size_t)B_ * D_ * 4);

    // ---- helpers ----
    auto tr = [&](const float* src, bf16_t* dst, int R, int C) {
        int n = R * C;
        k_transpose_bf16<<<(n + 255) / 256, 256, 0, stream>>>(src, dst, R, C);
    };
    auto gemm = [&](const bf16_t* A, const bf16_t* Wt, const float* bias, float* C,
                    bf16_t* Cb, int M, int N, int K, int accum) {
        // one wave per 16x64 tile; 4 waves (256 cols) per block
        dim3 blk(32, 4), grd((N + 255) / 256, M / 16);
        k_gemm_bf16<<<grd, blk, 0, stream>>>(A, Wt, bias, C, Cb, M, N, K, accum);
    };

    // ---- preprocessing: weight transposes to bf16 N-major ----
    tr(Wk, wk_t, TWOE_, D_);
    tr(Ws_init, wsinit_t, TWOE_, D_);
    tr(lg_Wx,   lgru_xw,               E_, 2048);
    tr(lg_Wlin, lgru_xw + 2048 * E_,   E_, 512);
    tr(lg_Wh,   lgru_hw,               D_, 1536);
    tr(lg_Whh,  lgru_hw + 1536 * D_,   D_, 512);
    for (int i = 0; i < 2; ++i) {
        tr(tg_Wh[i],  tgru_hw[i],              D_, 1024);
        tr(tg_Whh[i], tgru_hw[i] + 1024 * D_,  D_, 512);
        tr(ct_Wh[i],  ctgru_hw[i],             D_, 1024);
        tr(ct_Whh[i], ctgru_hw[i] + 1024 * D_, D_, 512);
    }
    tr(cl_Wx,   cond_xw,                 TWOE_, 2048);
    tr(cl_Wlin, cond_xw + 2048 * TWOE_,  TWOE_, 512);
    tr(cl_Wh,   cond_hw,                 D_, 1536);
    tr(cl_Whh,  cond_hw + 1536 * D_,     D_, 512);
    tr(a_Wq, wq_t, D_, D_);
    tr(a_Wo, wo_t, TWOE_, TWOE_);
    tr(Ws, ws_t, D_, D_);
    tr(Wy, wy_t, E_, D_);
    tr(Wc, wc_t, TWOE_, D_);

    // bias concatenations
    k_bias_cat<<<(2560 + 255) / 256, 256, 0, stream>>>(lg_b, lbias, 2048, 2560);
    k_bias_cat<<<(2560 + 255) / 256, 256, 0, stream>>>(cl_b, cbias, 2048, 2560);
    for (int i = 0; i < 2; ++i) {
        k_bias_cat<<<(1536 + 255) / 256, 256, 0, stream>>>(tg_b[i], tbias[i], 1024, 1536);
        k_bias_cat<<<(1536 + 255) / 256, 256, 0, stream>>>(ct_b[i], ctbias[i], 1024, 1536);
    }

    // activations to bf16
    {
        int n = B_ * L_ * TWOE_;
        k_convert_bf16<<<(n + 255) / 256, 256, 0, stream>>>(xs_h, xsb, n);
    }
    {
        int n = T_ * B_ * E_;
        k_embed<<<(n + 255) / 256, 256, 0, stream>>>(emb, ys, ysb);
    }

    // uh = xs_h @ Wk  (M = B*L, K = 2E, N = D)
    gemm(xsb, wk_t, nullptr, uh, nullptr, B_ * L_, D_, TWOE_, 0);

    // s0 = tanh(mean @ Ws_init + bs_init)
    k_mean<<<B_, 256, 0, stream>>>(xs_h, xs_mask, meanf, meanb);
    gemm(meanb, wsinit_t, bs_init, s0pre, nullptr, B_, D_, TWOE_, 0);
    k_tanh_state<<<(B_ * D_ + 255) / 256, 256, 0, stream>>>(s0pre, s_f, s_bf, B_ * D_);

    const int nBD = B_ * D_;
    const int gBD = (nBD + 255) / 256;

    // ---- recurrent scan ----
    for (int t = 0; t < T_; ++t) {
        const bf16_t* y_bf = ysb + (size_t)t * B_ * E_;

        // L-GRU (input = y)
        gemm(y_bf, lgru_xw, lbias, g1, nullptr, B_, 2560, E_, 0);
        gemm(s_bf, lgru_hw, nullptr, g2, nullptr, B_, 2048, D_, 0);
        k_lgru_combine<<<gBD, 256, 0, stream>>>(g1, g2, s_f, s_bf, ys_mask, t);

        // T-GRUs
        for (int i = 0; i < 2; ++i) {
            gemm(s_bf, tgru_hw[i], tbias[i], tg, nullptr, B_, 1536, D_, 0);
            k_tgru_combine<<<gBD, 256, 0, stream>>>(tg, tg_b[i] + 1024, s_f, s_bf, ys_mask, t);
        }

        // attention
        gemm(s_bf, wq_t, nullptr, qb, nullptr, B_, D_, D_, 0);
        k_attn<<<B_ * H_, 256, 0, stream>>>(qb, uh, xs_h, a_v, xs_mask, ctxpre, ctxpre_b);
        gemm(ctxpre_b, wo_t, nullptr, ctxf, ctx_b, B_, TWOE_, TWOE_, 0);

        // conditional L-GRU (input = ctx)
        gemm(ctx_b, cond_xw, cbias, g1, nullptr, B_, 2560, TWOE_, 0);
        gemm(s_bf, cond_hw, nullptr, g2, nullptr, B_, 2048, D_, 0);
        k_lgru_combine<<<gBD, 256, 0, stream>>>(g1, g2, s_f, s_bf, ys_mask, t);

        // conditional T-GRUs
        for (int i = 0; i < 2; ++i) {
            gemm(s_bf, ctgru_hw[i], ctbias[i], tg, nullptr, B_, 1536, D_, 0);
            k_tgru_combine<<<gBD, 256, 0, stream>>>(tg, ct_b[i] + 1024, s_f, s_bf, ys_mask, t);
        }

        // output: tanh(s@Ws + bs + y@Wy + by + ctx@Wc + bc) * mask
        gemm(s_bf,  ws_t, bs, osum, nullptr, B_, D_, D_, 0);
        gemm(y_bf,  wy_t, by, osum, nullptr, B_, D_, E_, 1);
        gemm(ctx_b, wc_t, bc, osum, nullptr, B_, D_, TWOE_, 1);
        k_out<<<gBD, 256, 0, stream>>>(osum, ys_mask, out, t);
    }
}
